// ProteinGNNEncoder_26199300506300
// MI455X (gfx1250) — matrix operand order repaired and checked by reading, not compiled, not run on hardware
//
#include <hip/hip_runtime.h>

#define NNODES 50000
#define NEDGES 1000000
#define HDIM   128
#define EDIM   4
#define NLAYERS 3
#define EPSF   1e-5f

typedef __attribute__((ext_vector_type(16))) _Float16 v16h;
typedef __attribute__((ext_vector_type(8)))  _Float16 v8h;
typedef __attribute__((ext_vector_type(8)))  float    v8f;

// ---------------------------------------------------------------------------
// D = A(16x32 f16) * B(32x16 f16) + C(16x16 f32)
__device__ __forceinline__ v8f wmma_f16(v16h a, v16h b, v8f c) {
  return __builtin_amdgcn_wmma_f32_16x16x32_f16(
      /*neg_a=*/false, a, /*neg_b=*/false, b,
      /*c_mod=*/(short)0, c, /*reuse_a=*/false, /*reuse_b=*/false);
}

// A-fragment (16x32, 16-bit) per ISA 7.12.2:
// lane<16 : row M=lane,    halfs = K[0..7],  K[16..23]
// lane>=16: row M=lane-16, halfs = K[8..15], K[24..31]
__device__ __forceinline__ v16h load_a_f16(const _Float16* base, int lane) {
  const int lo = (lane < 16) ? 0 : 8;
  union { v16h v; v8h h[2]; } u;
  u.h[0] = *(const v8h*)(base + lo);
  u.h[1] = *(const v8h*)(base + lo + 16);
  return u.v;
}

__device__ __forceinline__ v16h load_a_f32(const float* base, int lane) {
  const int lo = (lane < 16) ? 0 : 8;
  union { v16h v; _Float16 e[16]; } u;
#pragma unroll
  for (int i = 0; i < 8; ++i) {
    u.e[i]     = (_Float16)base[lo + i];
    u.e[i + 8] = (_Float16)base[lo + 16 + i];
  }
  return u.v;
}

// Pre-packed B fragment: 512 halfs per (ktile,ntile) fragment, lane-major.
__device__ __forceinline__ v16h load_b_frag(const _Float16* pk, int frag, int lane) {
  return *(const v16h*)(pk + (size_t)frag * 512 + lane * 16);
}

// Stage a contiguous global block into LDS with GLOBAL_LOAD_ASYNC_TO_LDS_B128
// (ASYNCcnt path: memory -> LDS, no VGPR round trip), then barrier.
// `bytes` must be a multiple of 16*blockDim.x.
__device__ __forceinline__ void stage_to_lds_async(const _Float16* gsrc,
                                                   _Float16* ldst, int bytes) {
  unsigned lbase = (unsigned)(uintptr_t)ldst;          // low 32 bits = LDS offset
  unsigned long long gbase = (unsigned long long)(uintptr_t)gsrc;
  for (int c = threadIdx.x * 16; c < bytes; c += blockDim.x * 16) {
    unsigned la = lbase + (unsigned)c;
    unsigned long long ga = gbase + (unsigned)c;
    asm volatile("global_load_async_to_lds_b128 %0, %1, off"
                 :: "v"(la), "v"(ga) : "memory");
  }
  asm volatile("s_wait_asynccnt 0x0" ::: "memory");
  __syncthreads();
}

// ---------------------------------------------------------------------------
// Pack W (Ksrc x 128, f32 row-major) into f16 WMMA-B fragment order.
// B layout (32x16, 16-bit): lane n<16 -> col N=n, K 0..7,16..23; lane n+16 -> col N=n, K 8..15,24..31.
__global__ void pack_b_kernel(const float* __restrict__ W, int Ksrc, int ktiles,
                              _Float16* __restrict__ dst) {
  int tid = blockIdx.x * blockDim.x + threadIdx.x;
  int total = ktiles * 8 * 512;
  if (tid >= total) return;
  int frag = tid >> 9;
  int rem  = tid & 511;
  int lane = rem >> 4;
  int i    = rem & 15;
  int kt = frag >> 3, nt = frag & 7;
  int n = nt * 16 + (lane & 15);
  int k = kt * 32 + ((lane < 16) ? 0 : 8) + ((i < 8) ? i : (i + 8));
  float v = (k < Ksrc) ? W[(size_t)k * HDIM + n] : 0.f;
  dst[tid] = (_Float16)v;
}

__global__ void gnn_zero_kernel(float* __restrict__ p, int n) {
  int t = blockIdx.x * blockDim.x + threadIdx.x;
  if (t < n) p[t] = 0.f;
}

// h = x @ proj_w + proj_b ; also produce f16 copy for WMMA A-operands.
__global__ void __launch_bounds__(256) gnn_proj_kernel(
    const float* __restrict__ x, const float* __restrict__ pw,
    const float* __restrict__ pb, float* __restrict__ hf32,
    _Float16* __restrict__ h16) {
  int tid = blockIdx.x * blockDim.x + threadIdx.x;
  if (tid >= NNODES * HDIM) return;
  int n = tid >> 7, j = tid & 127;
  float s = pb[j];
#pragma unroll
  for (int k = 0; k < 9; ++k) s += x[n * 9 + k] * pw[k * HDIM + j];
  hf32[tid] = s;
  h16[tid] = (_Float16)s;
}

// ---------------------------------------------------------------------------
// Per-edge message MLP + scatter-add.  One wave per 16-edge tile, grid-stride.
// W1|W2 fragments staged once per block into LDS (106496 B) via async copies.
__global__ void __launch_bounds__(256) gnn_edge_kernel(
    const _Float16* __restrict__ h16, const int* __restrict__ srcp,
    const int* __restrict__ dstp, const float* __restrict__ eattr,
    const _Float16* __restrict__ pW, const float* __restrict__ b1,
    const float* __restrict__ b2, float* __restrict__ agg) {
  __shared__ __align__(16) _Float16 sW[53248];          // W1 (36864) | W2 (16384)
  __shared__ __align__(16) _Float16 sMid[8][16 * 136];  // C->A relayout scratch
  __shared__ int sDst[8][16];
  __shared__ int sSrc[8][16];

  stage_to_lds_async(pW, sW, 53248 * 2);  // 6656 b128 chunks / 256 threads
  const _Float16* pW1 = sW;
  const _Float16* pW2 = sW + 36864;

  const int lane = threadIdx.x & 31;
  const int w    = threadIdx.x >> 5;
  const int gwave  = blockIdx.x * (blockDim.x >> 5) + w;
  const int nwaves = gridDim.x * (blockDim.x >> 5);
  const int ntiles = NEDGES / 16;
  const int row = lane & 15;
  const v8f vzero = {};

  for (int tile = gwave; tile < ntiles; tile += nwaves) {
    const int e0 = tile * 16;
    __builtin_prefetch(dstp + e0 + nwaves * 16, 0, 1);
    if (lane < 16) {
      sDst[w][lane] = dstp[e0 + lane];
      sSrc[w][lane] = srcp[e0 + lane];
    }
    asm volatile("s_wait_dscnt 0" ::: "memory");
    const _Float16* hD = h16 + (size_t)sDst[w][row] * HDIM;
    const _Float16* hS = h16 + (size_t)sSrc[w][row] * HDIM;

    v8f acc[8];
#pragma unroll
    for (int i = 0; i < 8; ++i) acc[i] = vzero;

    // GEMM 1: K = 288 (128 dst | 128 src | 4 edge_attr zero-padded to 32)
#pragma unroll
    for (int kt = 0; kt < 9; ++kt) {
      v16h a;
      if (kt < 4) {
        a = load_a_f16(hD + kt * 32, lane);
      } else if (kt < 8) {
        a = load_a_f16(hS + (kt - 4) * 32, lane);
      } else {
        union { v16h v; _Float16 e[16]; } u;
#pragma unroll
        for (int i = 0; i < 16; ++i) u.e[i] = (_Float16)0.f;
        if (lane < 16) {
          const float* ea = eattr + (size_t)(e0 + lane) * EDIM;
#pragma unroll
          for (int j = 0; j < EDIM; ++j) u.e[j] = (_Float16)ea[j];
        }
        a = u.v;
      }
#pragma unroll
      for (int nt = 0; nt < 8; ++nt)
        acc[nt] = wmma_f16(a, load_b_frag(pW1, kt * 8 + nt, lane), acc[nt]);
    }

    // bias + ReLU, relayout C-layout -> A-layout through LDS (wave-private)
    _Float16* sa = sMid[w];
#pragma unroll
    for (int nt = 0; nt < 8; ++nt) {
      const int col = nt * 16 + row;
      const float bias = b1[col];
#pragma unroll
      for (int r = 0; r < 8; ++r) {
        const int orow = (lane < 16) ? r : (8 + r);
        sa[orow * 136 + col] = (_Float16)fmaxf(acc[nt][r] + bias, 0.f);
      }
    }
    asm volatile("s_wait_dscnt 0" ::: "memory");

    // GEMM 2: K = 128
#pragma unroll
    for (int i = 0; i < 8; ++i) acc[i] = vzero;
#pragma unroll
    for (int kt = 0; kt < 4; ++kt) {
      v16h a = load_a_f16(sa + row * 136 + kt * 32, lane);
#pragma unroll
      for (int nt = 0; nt < 8; ++nt)
        acc[nt] = wmma_f16(a, load_b_frag(pW2, kt * 8 + nt, lane), acc[nt]);
    }

    // scatter-add (+b2) into agg by destination node (L2-resident atomics)
    int nd[8];
#pragma unroll
    for (int r = 0; r < 8; ++r) nd[r] = sDst[w][(lane < 16) ? r : (8 + r)];
#pragma unroll
    for (int nt = 0; nt < 8; ++nt) {
      const int col = nt * 16 + row;
      const float bias = b2[col];
#pragma unroll
      for (int r = 0; r < 8; ++r)
        atomicAdd(agg + (size_t)nd[r] * HDIM + col, acc[nt][r] + bias);
    }
  }
}

// ---------------------------------------------------------------------------
// Node update: h = LN(ReLU([h|agg]@U1+b1)@U2+b2 + h).  One wave per 16-node
// tile, grid-stride; U1|U2 fragments staged once per block into LDS (98304 B).
__global__ void __launch_bounds__(128) gnn_update_kernel(
    _Float16* __restrict__ h16, const float* __restrict__ agg,
    const _Float16* __restrict__ pU, const float* __restrict__ b1,
    const float* __restrict__ b2, const float* __restrict__ g,
    const float* __restrict__ bb, float* __restrict__ hf32) {
  __shared__ __align__(16) _Float16 sW[49152];          // U1 (32768) | U2 (16384)
  __shared__ __align__(16) _Float16 sMid[4][16 * 136];
  __shared__ float sF[4][16 * 132];   // stride 132 -> conflict-free row reads
  __shared__ float sMu[4][16];
  __shared__ float sRs[4][16];

  stage_to_lds_async(pU, sW, 49152 * 2);  // 6144 b128 chunks / 128 threads
  const _Float16* pU1 = sW;
  const _Float16* pU2 = sW + 32768;

  const int lane = threadIdx.x & 31;
  const int w    = threadIdx.x >> 5;
  const int gwave  = blockIdx.x * (blockDim.x >> 5) + w;
  const int nwaves = gridDim.x * (blockDim.x >> 5);
  const int ntiles = NNODES / 16;
  const int row  = lane & 15;
  const v8f vzero = {};

  for (int tile = gwave; tile < ntiles; tile += nwaves) {
    const int node = tile * 16 + row;

    v8f acc[8];
#pragma unroll
    for (int i = 0; i < 8; ++i) acc[i] = vzero;

    // GEMM 1: K = 256 (128 from h, 128 from agg converted on the fly)
#pragma unroll
    for (int kt = 0; kt < 8; ++kt) {
      v16h a = (kt < 4)
          ? load_a_f16(h16 + (size_t)node * HDIM + kt * 32, lane)
          : load_a_f32(agg + (size_t)node * HDIM + (kt - 4) * 32, lane);
#pragma unroll
      for (int nt = 0; nt < 8; ++nt)
        acc[nt] = wmma_f16(a, load_b_frag(pU1, kt * 8 + nt, lane), acc[nt]);
    }

    _Float16* sa = sMid[w];
#pragma unroll
    for (int nt = 0; nt < 8; ++nt) {
      const int col = nt * 16 + row;
      const float bias = b1[col];
#pragma unroll
      for (int r = 0; r < 8; ++r) {
        const int orow = (lane < 16) ? r : (8 + r);
        sa[orow * 136 + col] = (_Float16)fmaxf(acc[nt][r] + bias, 0.f);
      }
    }
    asm volatile("s_wait_dscnt 0" ::: "memory");

    // GEMM 2: K = 128
#pragma unroll
    for (int i = 0; i < 8; ++i) acc[i] = vzero;
#pragma unroll
    for (int kt = 0; kt < 4; ++kt) {
      v16h a = load_a_f16(sa + row * 136 + kt * 32, lane);
#pragma unroll
      for (int nt = 0; nt < 8; ++nt)
        acc[nt] = wmma_f16(a, load_b_frag(pU2, kt * 8 + nt, lane), acc[nt]);
    }

    // +b2 +residual -> LDS for layernorm statistics
    float* sf = sF[w];
#pragma unroll
    for (int nt = 0; nt < 8; ++nt) {
      const int col = nt * 16 + row;
      const float bias = b2[col];
#pragma unroll
      for (int r = 0; r < 8; ++r) {
        const int orow = (lane < 16) ? r : (8 + r);
        sf[orow * 132 + col] =
            acc[nt][r] + bias + hf32[(size_t)(tile * 16 + orow) * HDIM + col];
      }
    }
    asm volatile("s_wait_dscnt 0" ::: "memory");

    if (lane < 16) {
      float s = 0.f, s2 = 0.f;
      for (int j = 0; j < HDIM; ++j) {
        float v = sf[lane * 132 + j];
        s += v; s2 += v * v;
      }
      float mu  = s * (1.f / HDIM);
      float var = s2 * (1.f / HDIM) - mu * mu;
      sMu[w][lane] = mu;
      sRs[w][lane] = rsqrtf(var + EPSF);
    }
    asm volatile("s_wait_dscnt 0" ::: "memory");

#pragma unroll
    for (int nt = 0; nt < 8; ++nt) {
      const int col = nt * 16 + row;
      const float gg = g[col], bo = bb[col];
#pragma unroll
      for (int r = 0; r < 8; ++r) {
        const int orow = (lane < 16) ? r : (8 + r);
        const float y =
            (sf[orow * 132 + col] - sMu[w][orow]) * sRs[w][orow] * gg + bo;
        const size_t idx = (size_t)(tile * 16 + orow) * HDIM + col;
        hf32[idx] = y;
        h16[idx]  = (_Float16)y;
      }
    }
  }
}

// ---------------------------------------------------------------------------
extern "C" void kernel_launch(void* const* d_in, const int* in_sizes, int n_in,
                              void* d_out, int out_size, void* d_ws, size_t ws_size,
                              hipStream_t stream) {
  (void)in_sizes; (void)n_in; (void)out_size; (void)ws_size;
  const float* x      = (const float*)d_in[0];
  const int*   eidx   = (const int*)d_in[1];
  const float* eattr  = (const float*)d_in[2];
  const float* proj_w = (const float*)d_in[3];
  const float* proj_b = (const float*)d_in[4];
  const float* msg_w1 = (const float*)d_in[5];
  const float* msg_b1 = (const float*)d_in[6];
  const float* msg_w2 = (const float*)d_in[7];
  const float* msg_b2 = (const float*)d_in[8];
  const float* upd_w1 = (const float*)d_in[9];
  const float* upd_b1 = (const float*)d_in[10];
  const float* upd_w2 = (const float*)d_in[11];
  const float* upd_b2 = (const float*)d_in[12];
  const float* ln_g   = (const float*)d_in[13];
  const float* ln_b   = (const float*)d_in[14];
  const int* srcp = eidx;            // edge_index[0] = source
  const int* dstp = eidx + NEDGES;   // edge_index[1] = target

  float* hf32 = (float*)d_out;       // h lives in d_out across layers

  char* ws = (char*)d_ws;
  _Float16* h16    = (_Float16*)ws;                                   // 12.8 MB
  float*    agg    = (float*)(ws + (size_t)NNODES * HDIM * 2);        // 25.6 MB
  _Float16* packed = (_Float16*)(ws + (size_t)NNODES * HDIM * 6);     // 0.62 MB

  // per-layer packed layout (halfs): msgW1 [0,36864) msgW2 [36864,53248)
  //                                  updW1 [53248,86016) updW2 [86016,102400)
  const size_t PL = 102400;
  for (int l = 0; l < NLAYERS; ++l) {
    _Float16* base = packed + (size_t)l * PL;
    pack_b_kernel<<<(9 * 8 * 512 + 255) / 256, 256, 0, stream>>>(
        msg_w1 + (size_t)l * 260 * HDIM, 260, 9, base);
    pack_b_kernel<<<(4 * 8 * 512 + 255) / 256, 256, 0, stream>>>(
        msg_w2 + (size_t)l * HDIM * HDIM, 128, 4, base + 36864);
    pack_b_kernel<<<(8 * 8 * 512 + 255) / 256, 256, 0, stream>>>(
        upd_w1 + (size_t)l * 256 * HDIM, 256, 8, base + 53248);
    pack_b_kernel<<<(4 * 8 * 512 + 255) / 256, 256, 0, stream>>>(
        upd_w2 + (size_t)l * HDIM * HDIM, 128, 4, base + 86016);
  }

  gnn_proj_kernel<<<(NNODES * HDIM + 255) / 256, 256, 0, stream>>>(
      x, proj_w, proj_b, hf32, h16);

  for (int l = 0; l < NLAYERS; ++l) {
    _Float16* base = packed + (size_t)l * PL;
    gnn_zero_kernel<<<(NNODES * HDIM + 255) / 256, 256, 0, stream>>>(
        agg, NNODES * HDIM);
    gnn_edge_kernel<<<512, 256, 0, stream>>>(
        h16, srcp, dstp, eattr, base,
        msg_b1 + l * HDIM, msg_b2 + l * HDIM, agg);
    gnn_update_kernel<<<128, 128, 0, stream>>>(
        h16, agg, base + 53248,
        upd_b1 + l * HDIM, upd_b2 + l * HDIM,
        ln_g + l * HDIM, ln_b + l * HDIM, hf32);
  }
}
